// LSTM_68401649156117
// MI455X (gfx1250) — matrix operand order repaired and checked by reading, not compiled, not run on hardware
//
#include <hip/hip_runtime.h>
#include <hip/hip_bf16.h>

// ---------------------------------------------------------------------------
// LSTM for MI455X (gfx1250, wave32, WMMA), software-pipelined inner loops.
//
//   Phase A: convert X and the 4 weight matrices to bf16; split W into the
//            input part Wx [4H, E] and recurrent part Wh [4H, H]; zero h/c.
//   Phase B: one big parallel GEMM  pregate[t,b,g,h] = X @ Wx.T + bias
//            (fp32 accumulate via v_wmma_f32_16x16x32_bf16).
//   Phase C: 512 sequential per-timestep kernels. Each wave owns a 16x16
//            (batch x hidden) tile, accumulates ALL FOUR gate GEMMs
//            (h @ Wh.T) in registers, then applies the full LSTM pointwise
//            update in-register. h is ping-pong buffered in bf16; c stays fp32.
//
// K-loops are register double-buffered: fragments for step k+32 are loaded
// while the WMMAs for step k execute, hiding global load latency. WGP-scope
// (scope=0) global_prefetch_b8 via inline asm pulls the following K-step into
// the near cache (SE/SYS-scope prefetch would stop at GL2 per ISA 10.5).
//
// Workspace layout (assumes ws_size >= ~593 MB):
//   [pregate fp32 512MB][Xb bf16 64MB][Wxb bf16 8MB][Whb bf16 8MB]
//   [hbuf bf16 ping-pong 256KB][cbuf fp32 256KB]
// All prefetch addresses stay inside this contiguous allocation (safe for
// non-speculative WGP-scope prefetch).
// ---------------------------------------------------------------------------

typedef __attribute__((ext_vector_type(16))) __bf16 v16bf;
typedef __attribute__((ext_vector_type(8)))  __bf16 v8bf;
typedef __attribute__((ext_vector_type(8)))  float  v8f;

constexpr int T_ = 512;
constexpr int B_ = 64;
constexpr int E_ = 1024;
constexpr int H_ = 1024;

constexpr size_t PRE_BYTES = (size_t)T_ * B_ * 4 * H_ * sizeof(float);
constexpr size_t XB_OFF    = PRE_BYTES;
constexpr size_t XB_BYTES  = (size_t)T_ * B_ * E_ * 2;
constexpr size_t WX_OFF    = XB_OFF + XB_BYTES;
constexpr size_t WX_BYTES  = (size_t)4 * H_ * E_ * 2;
constexpr size_t WH_OFF    = WX_OFF + WX_BYTES;
constexpr size_t WH_BYTES  = (size_t)4 * H_ * H_ * 2;
constexpr size_t HB_OFF    = WH_OFF + WH_BYTES;
constexpr size_t HB_BYTES  = (size_t)2 * B_ * H_ * 2;   // ping-pong
constexpr size_t CB_OFF    = HB_OFF + HB_BYTES;

// --- Fragment loaders (contiguous k = 16*hi + s, same order for A and B) -------------
// A (16x32 bf16, row-major rows of the activation matrix):
//   lane L<16 : row M=L,     K = k0+0..15
//   lane L>=16: row M=L-16,  K = k0+16..31
__device__ __forceinline__ v16bf ld_a(const __bf16* rowp, int k0, int hi) {
  return *(const v16bf*)(rowp + k0 + hi * 16);
}
// B (32x16 bf16, columns are contiguous rows of W since we do inp @ W.T):
//   lane L<16 : col N=L,     K = k0+0..15   (one contiguous 32B chunk)
//   lane L>=16: col N=L-16,  K = k0+16..31
__device__ __forceinline__ v16bf ld_b(const __bf16* colp, int k0, int hi) {
  return *(const v16bf*)(colp + k0 + hi * 16);
}

// One pipeline stage: A fragment + B fragments for all four gates.
struct Frag5 { v16bf a, b0, b1, b2, b3; };

__device__ __forceinline__ Frag5 ld_frags(const __bf16* arow,
                                          const __bf16* b0, const __bf16* b1,
                                          const __bf16* b2, const __bf16* b3,
                                          int k0, int hi) {
  Frag5 f;
  f.a  = ld_a(arow, k0, hi);
  f.b0 = ld_b(b0, k0, hi);
  f.b1 = ld_b(b1, k0, hi);
  f.b2 = ld_b(b2, k0, hi);
  f.b3 = ld_b(b3, k0, hi);
  return f;
}

__device__ __forceinline__ float sigmoidf_(float x) {
  return 1.0f / (1.0f + __expf(-x));
}

#define WMMA_BF16(A, Bm, C) \
  __builtin_amdgcn_wmma_f32_16x16x32_bf16(false, (A), false, (Bm), (short)0, (C), false, false)

// WGP-scope prefetch (scope field 0): fills all cache levels down to WGP$,
// unlike __builtin_prefetch which only reaches SE/SYS scope (GL2-only).
__device__ __forceinline__ void prefetch_wgp(const void* p) {
  asm volatile("global_prefetch_b8 %0, off" :: "v"(p));
}

// --- Phase A: conversions ---------------------------------------------------
__global__ void lstm_cvt_x(const float* __restrict__ X, __bf16* __restrict__ Xb) {
  size_t i = ((size_t)blockIdx.x * blockDim.x + threadIdx.x) * 4;
  float4 v = *(const float4*)(X + i);
  Xb[i + 0] = (__bf16)v.x;
  Xb[i + 1] = (__bf16)v.y;
  Xb[i + 2] = (__bf16)v.z;
  Xb[i + 3] = (__bf16)v.w;
}

__global__ void lstm_pack_w(const float* __restrict__ Wf, const float* __restrict__ Wi,
                            const float* __restrict__ Wo, const float* __restrict__ Wc,
                            __bf16* __restrict__ Wxb, __bf16* __restrict__ Whb) {
  const size_t per = (size_t)H_ * (E_ + H_);
  size_t idx = (size_t)blockIdx.x * blockDim.x + threadIdx.x;
  if (idx >= 4 * per) return;
  int    g   = (int)(idx / per);
  size_t rem = idx - (size_t)g * per;
  int    h   = (int)(rem / (E_ + H_));
  int    col = (int)(rem - (size_t)h * (E_ + H_));
  const float* W = (g == 0) ? Wf : (g == 1) ? Wi : (g == 2) ? Wo : Wc;
  float v = W[rem];
  if (col < E_) Wxb[((size_t)g * H_ + h) * E_ + col]        = (__bf16)v;
  else          Whb[((size_t)g * H_ + h) * H_ + (col - E_)] = (__bf16)v;
}

__global__ void lstm_init_state(__bf16* __restrict__ hbuf, float* __restrict__ cbuf) {
  int i = blockIdx.x * blockDim.x + threadIdx.x;   // B_*H_ threads
  hbuf[i]           = (__bf16)0.0f;
  hbuf[B_ * H_ + i] = (__bf16)0.0f;
  cbuf[i]           = 0.0f;
}

// --- Phase B: pregate = X @ Wx.T + bias (all 4 gates per wave) --------------
__global__ void __launch_bounds__(256)
lstm_xgemm(const __bf16* __restrict__ Xb,      // [T*B, E]
           const __bf16* __restrict__ Wxb,     // [4H, E]
           const float* __restrict__ bias0, const float* __restrict__ bias1,
           const float* __restrict__ bias2, const float* __restrict__ bias3,
           float* __restrict__ pregate)        // [T*B, 4, H]
{
  const int wave   = blockIdx.x * (blockDim.x >> 5) + (threadIdx.x >> 5);
  const int nTiles = H_ / 16;                  // 64
  const int mt     = wave / nTiles;            // 0 .. T*B/16-1
  const int nt     = wave - mt * nTiles;       // 0 .. 63
  const int lane   = threadIdx.x & 31;
  const int r      = lane & 15;
  const int hi     = lane >> 4;

  const __bf16* arow  = Xb + (size_t)(mt * 16 + r) * E_;
  const __bf16* bcol0 = Wxb + (size_t)(0 * H_ + nt * 16 + r) * E_;
  const __bf16* bcol1 = Wxb + (size_t)(1 * H_ + nt * 16 + r) * E_;
  const __bf16* bcol2 = Wxb + (size_t)(2 * H_ + nt * 16 + r) * E_;
  const __bf16* bcol3 = Wxb + (size_t)(3 * H_ + nt * 16 + r) * E_;

  v8f acc0 = {}, acc1 = {}, acc2 = {}, acc3 = {};

  // Software pipeline: load k0 fragments, then overlap k0+32 loads with WMMAs.
  Frag5 cur = ld_frags(arow, bcol0, bcol1, bcol2, bcol3, 0, hi);
  for (int k0 = 0; k0 < E_ - 32; k0 += 32) {
    const int kn = k0 + 32;
    prefetch_wgp(arow + kn + 32);
    prefetch_wgp(bcol0 + kn + 32);
    prefetch_wgp(bcol1 + kn + 32);
    prefetch_wgp(bcol2 + kn + 32);
    prefetch_wgp(bcol3 + kn + 32);
    Frag5 nxt = ld_frags(arow, bcol0, bcol1, bcol2, bcol3, kn, hi);
    acc0 = WMMA_BF16(cur.a, cur.b0, acc0);
    acc1 = WMMA_BF16(cur.a, cur.b1, acc1);
    acc2 = WMMA_BF16(cur.a, cur.b2, acc2);
    acc3 = WMMA_BF16(cur.a, cur.b3, acc3);
    cur = nxt;
  }
  acc0 = WMMA_BF16(cur.a, cur.b0, acc0);
  acc1 = WMMA_BF16(cur.a, cur.b1, acc1);
  acc2 = WMMA_BF16(cur.a, cur.b2, acc2);
  acc3 = WMMA_BF16(cur.a, cur.b3, acc3);

  // C/D layout: element e of acc -> (m = e + 8*hi, n = lane&15) within tile
  const int   n  = nt * 16 + r;
  const int   mb = mt * 16 + hi * 8;
  const float bb0 = bias0[n], bb1 = bias1[n], bb2 = bias2[n], bb3 = bias3[n];
#pragma unroll
  for (int e = 0; e < 8; ++e) {
    size_t row4 = (size_t)(mb + e) * 4;
    pregate[(row4 + 0) * H_ + n] = acc0[e] + bb0;
    pregate[(row4 + 1) * H_ + n] = acc1[e] + bb1;
    pregate[(row4 + 2) * H_ + n] = acc2[e] + bb2;
    pregate[(row4 + 3) * H_ + n] = acc3[e] + bb3;
  }
}

// --- Phase C: one fused recurrent step (h @ Wh.T + pregate -> gates -> h,c) -
__global__ void __launch_bounds__(128)
lstm_step(int t,
          const __bf16* __restrict__ Whb,       // [4H, H]
          const float*  __restrict__ pregate,   // [T*B, 4, H]
          __bf16*       __restrict__ hbuf,      // [2][B, H] ping-pong
          float*        __restrict__ cbuf,      // [B, H]
          float*        __restrict__ out)       // [B, T, H]
{
  const int wave   = blockIdx.x * (blockDim.x >> 5) + (threadIdx.x >> 5);
  const int nTiles = H_ / 16;                   // 64
  const int mt     = wave / nTiles;             // 0..3  (batch tiles)
  const int nt     = wave - mt * nTiles;        // 0..63 (hidden tiles)
  const int lane   = threadIdx.x & 31;
  const int r      = lane & 15;
  const int hi     = lane >> 4;

  const __bf16* hcur  = hbuf + (size_t)(t & 1) * B_ * H_;
  __bf16*       hnext = hbuf + (size_t)((t + 1) & 1) * B_ * H_;

  const __bf16* arow  = hcur + (size_t)(mt * 16 + r) * H_;
  const __bf16* bcol0 = Whb + (size_t)(0 * H_ + nt * 16 + r) * H_;
  const __bf16* bcol1 = Whb + (size_t)(1 * H_ + nt * 16 + r) * H_;
  const __bf16* bcol2 = Whb + (size_t)(2 * H_ + nt * 16 + r) * H_;
  const __bf16* bcol3 = Whb + (size_t)(3 * H_ + nt * 16 + r) * H_;

  v8f acc0 = {}, acc1 = {}, acc2 = {}, acc3 = {};

  Frag5 cur = ld_frags(arow, bcol0, bcol1, bcol2, bcol3, 0, hi);
  for (int k0 = 0; k0 < H_ - 32; k0 += 32) {
    const int kn = k0 + 32;
    prefetch_wgp(arow + kn + 32);
    prefetch_wgp(bcol0 + kn + 32);
    prefetch_wgp(bcol1 + kn + 32);
    prefetch_wgp(bcol2 + kn + 32);
    prefetch_wgp(bcol3 + kn + 32);
    Frag5 nxt = ld_frags(arow, bcol0, bcol1, bcol2, bcol3, kn, hi);
    acc0 = WMMA_BF16(cur.a, cur.b0, acc0);
    acc1 = WMMA_BF16(cur.a, cur.b1, acc1);
    acc2 = WMMA_BF16(cur.a, cur.b2, acc2);
    acc3 = WMMA_BF16(cur.a, cur.b3, acc3);
    cur = nxt;
  }
  acc0 = WMMA_BF16(cur.a, cur.b0, acc0);
  acc1 = WMMA_BF16(cur.a, cur.b1, acc1);
  acc2 = WMMA_BF16(cur.a, cur.b2, acc2);
  acc3 = WMMA_BF16(cur.a, cur.b3, acc3);

  const int n  = nt * 16 + r;
  const int mb = mt * 16 + hi * 8;
  const float* pg = pregate + (size_t)t * B_ * 4 * H_;

#pragma unroll
  for (int e = 0; e < 8; ++e) {
    const int b = mb + e;                        // batch row 0..63
    size_t row4 = (size_t)b * 4;
    float pf = acc0[e] + pg[(row4 + 0) * H_ + n];
    float pi = acc1[e] + pg[(row4 + 1) * H_ + n];
    float po = acc2[e] + pg[(row4 + 2) * H_ + n];
    float pc = acc3[e] + pg[(row4 + 3) * H_ + n];

    float f  = sigmoidf_(pf);
    float i  = sigmoidf_(pi);
    float o  = sigmoidf_(po);
    float cg = tanhf(pc);

    size_t bh = (size_t)b * H_ + n;
    float cn = f * cbuf[bh] + i * cg;
    cbuf[bh] = cn;
    float hn = o * tanhf(cn);
    hnext[bh] = (__bf16)hn;
    out[((size_t)b * T_ + t) * H_ + n] = hn;     // out is [B, T, H]
  }
}

// ---------------------------------------------------------------------------
extern "C" void kernel_launch(void* const* d_in, const int* in_sizes, int n_in,
                              void* d_out, int out_size, void* d_ws, size_t ws_size,
                              hipStream_t stream) {
  (void)in_sizes; (void)n_in; (void)out_size; (void)ws_size;

  const float* X   = (const float*)d_in[0];
  const float* Wf  = (const float*)d_in[1];
  const float* bf_ = (const float*)d_in[2];
  const float* Wi  = (const float*)d_in[3];
  const float* bi_ = (const float*)d_in[4];
  const float* Wo  = (const float*)d_in[5];
  const float* bo_ = (const float*)d_in[6];
  const float* Wc  = (const float*)d_in[7];
  const float* bc_ = (const float*)d_in[8];
  float* out = (float*)d_out;

  char* ws = (char*)d_ws;
  float*  pregate = (float*)ws;
  __bf16* Xb      = (__bf16*)(ws + XB_OFF);
  __bf16* Wxb     = (__bf16*)(ws + WX_OFF);
  __bf16* Whb     = (__bf16*)(ws + WH_OFF);
  __bf16* hbuf    = (__bf16*)(ws + HB_OFF);
  float*  cbuf    = (float*)(ws + CB_OFF);

  // Phase A: conversions + state init
  {
    size_t nx = (size_t)T_ * B_ * E_;                 // 33,554,432 (div by 1024)
    lstm_cvt_x<<<(unsigned)(nx / (256 * 4)), 256, 0, stream>>>(X, Xb);

    size_t nw = (size_t)4 * H_ * (E_ + H_);           // 8,388,608
    lstm_pack_w<<<(unsigned)((nw + 255) / 256), 256, 0, stream>>>(Wf, Wi, Wo, Wc, Wxb, Whb);

    lstm_init_state<<<(B_ * H_) / 256, 256, 0, stream>>>(hbuf, cbuf);
  }

  // Phase B: big parallel input-projection GEMM (all timesteps at once)
  {
    int waves  = (T_ * B_ / 16) * (H_ / 16);          // 2048 * 64 = 131072
    int blocks = waves / 8;                           // 8 waves (256 thr) per WG
    lstm_xgemm<<<blocks, 256, 0, stream>>>(Xb, Wxb, bf_, bi_, bo_, bc_, pregate);
  }

  // Phase C: sequential recurrent steps (256 waves, 4 waves per WG)
  for (int t = 0; t < T_; ++t) {
    lstm_step<<<64, 128, 0, stream>>>(t, Whb, pregate, hbuf, cbuf, out);
  }
}
